// SileroVAD_66821101191583
// MI455X (gfx1250) — matrix operand order repaired and checked by reading, not compiled, not run on hardware
//
#include <hip/hip_runtime.h>
#include <hip/hip_bf16.h>
#include <math.h>

typedef __attribute__((ext_vector_type(16))) __bf16 v16bf;
typedef __attribute__((ext_vector_type(8)))  __bf16 v8bf;
typedef __attribute__((ext_vector_type(8)))  float  v8f;

#define ROWS        64
#define WIN_STRIDE  648                      // halves/row: 640-sample window + pad
#define MAG_STRIDE  168                      // 129 channels padded to 168 (mult of 8)
#define MAG_FRAME   (ROWS * MAG_STRIDE)
#define E1_STRIDE   136
#define E1_FRAME    (ROWS * E1_STRIDE)
#define E2_STRIDE   72
#define E2_FRAME    (ROWS * E2_STRIDE)
#define E3_STRIDE   72
#define E4_STRIDE   136
#define H0_OFF_H    8192                     // halves offset inside WIN region
#define GATE_STRIDE 520                      // f32 stride for 512 gates

#define WIN_OFF_B   0
#define MAG_OFF_B   (ROWS * WIN_STRIDE * 2)                 //  82944
#define GATES_OFF_B (MAG_OFF_B + 4 * MAG_FRAME * 2)         // 168960
#define HSUM_OFF_B  (GATES_OFF_B + ROWS * GATE_STRIDE * 4)  // 302080
#define SMEM_BYTES  (HSUM_OFF_B + ROWS * 4)                 // 302336 (< 320 KB)

// ---------------------------------------------------------------- fragments
// A fragment 16x32 bf16 from row-major LDS tile. base = &A[row0][k0].
__device__ __forceinline__ v16bf frag_a_lds(const __bf16* base, int strideH, int lane) {
  const __bf16* p = base + (lane & 15) * strideH + ((lane >> 4) << 3);
  v8bf lo = *reinterpret_cast<const v8bf*>(p);
  v8bf hi = *reinterpret_cast<const v8bf*>(p + 16);
  v16bf r;
#pragma unroll
  for (int i = 0; i < 8; ++i) { r[i] = lo[i]; r[i + 8] = hi[i]; }
  return r;
}

// B fragment 32x16 bf16 from B^T row-major in global (L2-hot weights).
// base = &Bt[n0][k0], ldk = K stride of Bt rows.
__device__ __forceinline__ v16bf frag_b_glb(const __bf16* base, int ldk, int lane) {
  const __bf16* p = base + (lane & 15) * ldk + ((lane >> 4) << 3);
  v8bf lo = *reinterpret_cast<const v8bf*>(p);
  v8bf hi = *reinterpret_cast<const v8bf*>(p + 16);
  v16bf r;
#pragma unroll
  for (int i = 0; i < 8; ++i) { r[i] = lo[i]; r[i + 8] = hi[i]; }
  return r;
}

__device__ __forceinline__ v8f wmma_bf16(v16bf a, v16bf b, v8f c) {
  return __builtin_amdgcn_wmma_f32_16x16x32_bf16(false, a, false, b, (short)0, c,
                                                 false, false);
}

__device__ __forceinline__ float sigf(float v) { return 1.f / (1.f + __expf(-v)); }

// ---------------------------------------------------------------- weight prep
// dst[n*Kd + k] = (n<Ns && k<Ks) ? src[n*rowStride + k*inner + off] : 0   (bf16)
__global__ void vad_cvt_pad(const float* __restrict__ src, __bf16* __restrict__ dst,
                            int Nd, int Kd, int Ns, int Ks, int rowStride, int inner,
                            int off) {
  int i = blockIdx.x * blockDim.x + threadIdx.x;
  if (i >= Nd * Kd) return;
  int n = i / Kd, k = i % Kd;
  float v = (n < Ns && k < Ks) ? src[n * rowStride + k * inner + off] : 0.f;
  dst[i] = (__bf16)v;
}

// ---------------------------------------------------------------- fused main
__global__ __launch_bounds__(256, 1) void vad_main(
    const float* __restrict__ x, const float* __restrict__ h0,
    const float* __restrict__ c0, const float* __restrict__ b1,
    const float* __restrict__ b2, const float* __restrict__ b3,
    const float* __restrict__ b4, const float* __restrict__ b_ih,
    const float* __restrict__ b_hh, const float* __restrict__ dec_w,
    const float* __restrict__ dec_b, const __bf16* __restrict__ Wstft,
    const __bf16* __restrict__ W1, const __bf16* __restrict__ W2,
    const __bf16* __restrict__ W3, const __bf16* __restrict__ W4,
    const __bf16* __restrict__ Wih, const __bf16* __restrict__ Whh,
    float* __restrict__ out, int B) {
  extern __shared__ char smem[];
  __bf16* winb  = (__bf16*)(smem + WIN_OFF_B);   // reflect-padded window (bf16)
  __bf16* magb  = (__bf16*)(smem + MAG_OFF_B);   // |STFT| [frame][row][ch]
  __bf16* e1b   = winb;                          // reuse WIN after phase 1
  __bf16* e2b   = magb;                          // reuse MAG after phase 2
  __bf16* e3b   = winb;                          // reuse after phase 3
  __bf16* h0b   = winb + H0_OFF_H;
  __bf16* e4b   = magb;                          // reuse after phase 4a
  float*  gates = (float*)(smem + GATES_OFF_B);
  float*  hsum  = (float*)(smem + HSUM_OFF_B);

  const int  tid  = threadIdx.x;
  const int  wave = tid >> 5;
  const int  lane = tid & 31;
  const int  col16 = lane & 15;
  const int  hi8   = (lane >> 4) << 3;
  const long row0  = (long)blockIdx.x * ROWS;

  // ---- phase 0: build reflect-padded window (bf16) + zero mag pad --------
  // xp[0:640] per row = [x0, 0*127, x[0:512]]   (frames t use [128t,128t+256))
  for (int i = tid; i < ROWS * WIN_STRIDE; i += 256) {
    int r = i / WIN_STRIDE, c = i % WIN_STRIDE;
    float v = 0.f;
    const float* xr = x + (row0 + r) * 512;
    if (c == 0) v = xr[0];
    else if (c >= 128 && c < 640) v = xr[c - 128];
    winb[i] = (__bf16)v;
  }
  for (int i = tid; i < 4 * MAG_FRAME; i += 256) magb[i] = (__bf16)0.f;
  __syncthreads();

  // ---- phase 1: STFT (real+imag tiles) -> magnitude ----------------------
  for (int j = wave; j < 144; j += 8) {
    int t = j / 36, r = j % 36, mt = r / 9, nt = r % 9;
    const __bf16* ab = winb + (mt * 16) * WIN_STRIDE + t * 128;
    v8f accR = {}, accI = {};
#pragma unroll
    for (int kb = 0; kb < 8; ++kb) {
      v16bf a = frag_a_lds(ab + kb * 32, WIN_STRIDE, lane);
      accR = wmma_bf16(a, frag_b_glb(Wstft + (nt * 16) * 256 + kb * 32, 256, lane), accR);
      accI = wmma_bf16(a, frag_b_glb(Wstft + (nt * 16 + 129) * 256 + kb * 32, 256, lane), accI);
    }
    int ch = nt * 16 + col16;
    if (ch < 129) {
      __bf16* d = magb + t * MAG_FRAME + (mt * 16) * MAG_STRIDE + ch;
#pragma unroll
      for (int v = 0; v < 8; ++v)
        d[(v + hi8) * MAG_STRIDE] = (__bf16)sqrtf(accR[v] * accR[v] + accI[v] * accI[v]);
    }
  }
  __syncthreads();

  // ---- phase 2: conv1 (K=129 padded 160) + ReLU -> e1[4][64][128] --------
  for (int j = wave; j < 128; j += 8) {
    int t = j >> 5, r = j & 31, mt = r >> 3, nt = r & 7;
    v8f acc = {};
#pragma unroll
    for (int tau = 0; tau < 3; ++tau) {
      int ts = t - 1 + tau;
      if (ts < 0 || ts >= 4) continue;
      const __bf16* ab = magb + ts * MAG_FRAME + (mt * 16) * MAG_STRIDE;
      const __bf16* wb = W1 + tau * (128 * 160) + (nt * 16) * 160;
#pragma unroll
      for (int kb = 0; kb < 5; ++kb)
        acc = wmma_bf16(frag_a_lds(ab + kb * 32, MAG_STRIDE, lane),
                        frag_b_glb(wb + kb * 32, 160, lane), acc);
    }
    int n = nt * 16 + col16;
    float bias = b1[n];
    __bf16* d = e1b + t * E1_FRAME + (mt * 16) * E1_STRIDE + n;
#pragma unroll
    for (int v = 0; v < 8; ++v) d[(v + hi8) * E1_STRIDE] = (__bf16)fmaxf(acc[v] + bias, 0.f);
  }
  __syncthreads();

  // ---- phase 3: conv2 stride2 (K=128) + ReLU -> e2[2][64][64] ------------
  for (int j = wave; j < 32; j += 8) {
    int t = j >> 4, r = j & 15, mt = r >> 2, nt = r & 3;
    v8f acc = {};
#pragma unroll
    for (int tau = 0; tau < 3; ++tau) {
      int ts = 2 * t - 1 + tau;
      if (ts < 0 || ts >= 4) continue;
      const __bf16* ab = e1b + ts * E1_FRAME + (mt * 16) * E1_STRIDE;
      const __bf16* wb = W2 + tau * (64 * 128) + (nt * 16) * 128;
#pragma unroll
      for (int kb = 0; kb < 4; ++kb)
        acc = wmma_bf16(frag_a_lds(ab + kb * 32, E1_STRIDE, lane),
                        frag_b_glb(wb + kb * 32, 128, lane), acc);
    }
    int n = nt * 16 + col16;
    float bias = b2[n];
    __bf16* d = e2b + t * E2_FRAME + (mt * 16) * E2_STRIDE + n;
#pragma unroll
    for (int v = 0; v < 8; ++v) d[(v + hi8) * E2_STRIDE] = (__bf16)fmaxf(acc[v] + bias, 0.f);
  }
  __syncthreads();

  // ---- phase 4a: conv3 stride2 (K=64) + ReLU -> e3[64][64] ---------------
  for (int j = wave; j < 16; j += 8) {
    int mt = j >> 2, nt = j & 3;
    v8f acc = {};
#pragma unroll
    for (int tau = 1; tau < 3; ++tau) {           // taps 0 falls off (pad)
      const __bf16* ab = e2b + (tau - 1) * E2_FRAME + (mt * 16) * E2_STRIDE;
      const __bf16* wb = W3 + tau * (64 * 64) + (nt * 16) * 64;
#pragma unroll
      for (int kb = 0; kb < 2; ++kb)
        acc = wmma_bf16(frag_a_lds(ab + kb * 32, E2_STRIDE, lane),
                        frag_b_glb(wb + kb * 32, 64, lane), acc);
    }
    int n = nt * 16 + col16;
    float bias = b3[n];
    __bf16* d = e3b + (mt * 16) * E3_STRIDE + n;
#pragma unroll
    for (int v = 0; v < 8; ++v) d[(v + hi8) * E3_STRIDE] = (__bf16)fmaxf(acc[v] + bias, 0.f);
  }
  __syncthreads();

  // ---- phase 4b: conv4 (K=64, only center tap valid) -> e4[64][128] ------
  for (int j = wave; j < 32; j += 8) {
    int mt = j >> 3, nt = j & 7;
    v8f acc = {};
    const __bf16* ab = e3b + (mt * 16) * E3_STRIDE;
    const __bf16* wb = W4 + 1 * (128 * 64) + (nt * 16) * 64;
#pragma unroll
    for (int kb = 0; kb < 2; ++kb)
      acc = wmma_bf16(frag_a_lds(ab + kb * 32, E3_STRIDE, lane),
                      frag_b_glb(wb + kb * 32, 64, lane), acc);
    int n = nt * 16 + col16;
    float bias = b4[n];
    __bf16* d = e4b + (mt * 16) * E4_STRIDE + n;
#pragma unroll
    for (int v = 0; v < 8; ++v) d[(v + hi8) * E4_STRIDE] = (__bf16)fmaxf(acc[v] + bias, 0.f);
  }
  __syncthreads();

  // ---- phase 5a: stage h0 (f32->bf16) into LDS; zero decoder sums --------
  for (int i = tid; i < ROWS * 128; i += 256) {
    int r = i >> 7, u = i & 127;
    h0b[r * E4_STRIDE + u] = (__bf16)h0[(row0 + r) * 128 + u];
  }
  if (tid < ROWS) hsum[tid] = 0.f;
  __syncthreads();

  // ---- phase 5b: LSTM gates = e4*Wih^T + h0*Whh^T + biases ---------------
  for (int j = wave; j < 128; j += 8) {
    int mt = j >> 5, nt = j & 31;
    int n = nt * 16 + col16;
    float bias = b_ih[n] + b_hh[n];
    v8f acc;
#pragma unroll
    for (int v = 0; v < 8; ++v) acc[v] = bias;
#pragma unroll
    for (int kb = 0; kb < 4; ++kb) {
      acc = wmma_bf16(frag_a_lds(e4b + (mt * 16) * E4_STRIDE + kb * 32, E4_STRIDE, lane),
                      frag_b_glb(Wih + n * 0 + (nt * 16) * 128 + kb * 32, 128, lane), acc);
      acc = wmma_bf16(frag_a_lds(h0b + (mt * 16) * E4_STRIDE + kb * 32, E4_STRIDE, lane),
                      frag_b_glb(Whh + (nt * 16) * 128 + kb * 32, 128, lane), acc);
    }
    float* g = gates + (mt * 16) * GATE_STRIDE + n;
#pragma unroll
    for (int v = 0; v < 8; ++v) g[(v + hi8) * GATE_STRIDE] = acc[v];
  }
  __syncthreads();

  // ---- phase 6: LSTM nonlinearity + decoder ------------------------------
  float* outH = out + B;
  float* outC = out + B + (long)B * 128;
  float dbias = dec_b[0];
  for (int i = tid; i < ROWS * 128; i += 256) {
    int r = i >> 7, u = i & 127;
    const float* g = gates + r * GATE_STRIDE;
    float ig = g[u], fg = g[128 + u], gg = g[256 + u], og = g[384 + u];
    float c0v = c0[(row0 + r) * 128 + u];
    float c1 = sigf(fg) * c0v + sigf(ig) * tanhf(gg);
    float h1 = sigf(og) * tanhf(c1);
    outH[(row0 + r) * 128 + u] = h1;
    outC[(row0 + r) * 128 + u] = c1;
    atomicAdd(&hsum[r], fmaxf(h1, 0.f) * dec_w[u]);
  }
  __syncthreads();
  if (tid < ROWS) out[row0 + tid] = sigf(hsum[tid] + dbias);
}

// ---------------------------------------------------------------- launcher
extern "C" void kernel_launch(void* const* d_in, const int* in_sizes, int n_in,
                              void* d_out, int out_size, void* d_ws, size_t ws_size,
                              hipStream_t stream) {
  const float* x     = (const float*)d_in[0];
  const float* h0    = (const float*)d_in[1];
  const float* c0    = (const float*)d_in[2];
  const float* basis = (const float*)d_in[3];
  const float* w1    = (const float*)d_in[4];
  const float* b1    = (const float*)d_in[5];
  const float* w2    = (const float*)d_in[6];
  const float* b2    = (const float*)d_in[7];
  const float* w3    = (const float*)d_in[8];
  const float* b3    = (const float*)d_in[9];
  const float* w4    = (const float*)d_in[10];
  const float* b4    = (const float*)d_in[11];
  const float* w_ih  = (const float*)d_in[12];
  const float* w_hh  = (const float*)d_in[13];
  const float* b_ih  = (const float*)d_in[14];
  const float* b_hh  = (const float*)d_in[15];
  const float* dec_w = (const float*)d_in[16];
  const float* dec_b = (const float*)d_in[17];
  const int B = in_sizes[0] / 512;

  // bf16 weight arena in d_ws (padded, B^T row-major)
  __bf16* Wstft = (__bf16*)d_ws;              // 288 x 256 (129 re + 129 im + pad)
  __bf16* W1p   = Wstft + 288 * 256;          // 3 x [128 x 160]
  __bf16* W2p   = W1p + 3 * 128 * 160;        // 3 x [64 x 128]
  __bf16* W3p   = W2p + 3 * 64 * 128;         // 3 x [64 x 64]
  __bf16* W4p   = W3p + 3 * 64 * 64;          // 3 x [128 x 64]
  __bf16* Wih   = W4p + 3 * 128 * 64;         // 512 x 128
  __bf16* Whh   = Wih + 512 * 128;            // 512 x 128

  auto cvt = [&](const float* s, __bf16* d, int Nd, int Kd, int Ns, int Ks, int rs,
                 int inner, int off) {
    int total = Nd * Kd;
    vad_cvt_pad<<<(total + 255) / 256, 256, 0, stream>>>(s, d, Nd, Kd, Ns, Ks, rs,
                                                         inner, off);
  };
  cvt(basis, Wstft, 288, 256, 258, 256, 256, 1, 0);
  for (int t = 0; t < 3; ++t) cvt(w1, W1p + t * 128 * 160, 128, 160, 128, 129, 129 * 3, 3, t);
  for (int t = 0; t < 3; ++t) cvt(w2, W2p + t * 64 * 128, 64, 128, 64, 128, 128 * 3, 3, t);
  for (int t = 0; t < 3; ++t) cvt(w3, W3p + t * 64 * 64, 64, 64, 64, 64, 64 * 3, 3, t);
  for (int t = 0; t < 3; ++t) cvt(w4, W4p + t * 128 * 64, 128, 64, 128, 64, 64 * 3, 3, t);
  cvt(w_ih, Wih, 512, 128, 512, 128, 128, 1, 0);
  cvt(w_hh, Whh, 512, 128, 512, 128, 128, 1, 0);

  (void)hipFuncSetAttribute((const void*)vad_main,
                            hipFuncAttributeMaxDynamicSharedMemorySize, SMEM_BYTES);
  vad_main<<<B / ROWS, 256, SMEM_BYTES, stream>>>(
      x, h0, c0, b1, b2, b3, b4, b_ih, b_hh, dec_w, dec_b, Wstft, W1p, W2p, W3p,
      W4p, Wih, Whh, (float*)d_out, B);
}
